// StockPredictorNet_70858370449602
// MI455X (gfx1250) — matrix operand order repaired and checked by reading, not compiled, not run on hardware
//
#include <hip/hip_runtime.h>
#include <hip/hip_bf16.h>

#define BATCH 4096
#define SEQ   256
#define IND   32
#define HID   64

typedef __attribute__((ext_vector_type(16))) _Float16 v16h;
typedef __attribute__((ext_vector_type(8)))  _Float16 v8h;
typedef __attribute__((ext_vector_type(4)))  _Float16 v4h;
typedef __attribute__((ext_vector_type(2)))  _Float16 v2h;
typedef __attribute__((ext_vector_type(8)))  float    v8f;
typedef __attribute__((ext_vector_type(4)))  float    v4f;
typedef __attribute__((ext_vector_type(2)))  float    v2f;

// ---- workspace layout (halves, then floats) ----
#define PK0X_OFF 0          // layer0 x-part : 16 tiles * 512 = 8192 halves
#define PK0H_OFF 8192       // layer0 h-part : 32 tiles * 512 = 16384 halves
#define PK1X_OFF 24576      // layer1 in-part: 16384 halves
#define PK1H_OFF 40960      // layer1 h-part : 16384 halves
#define WS_HALVES 57344
#define BSUM0_BYTE (WS_HALVES * 2)        // 256 f32: bih0+bhh0
#define BSUM1_BYTE (BSUM0_BYTE + 1024)    // 256 f32: bih1+bhh1

#define LOG2E 1.44269504088896340736f

#define WMMA_F16(A, B, C) \
  __builtin_amdgcn_wmma_f32_16x16x32_f16(false, (A), false, (B), (short)0, (C), false, false)

// ---------------------------------------------------------------------------
// Prep: transpose + convert weights into the exact WMMA-B register image.
// B tile (32x16 f16): lane l holds col n = l&15, K = (l>>4)*16 + j (j=0..15),
// packed so lane l reads halves [16*l .. 16*l+15] contiguously (32B).
// ---------------------------------------------------------------------------
__device__ inline void pack_w(const float* __restrict__ W, int K,
                              _Float16* __restrict__ dst, int tid, int nthr) {
  const int ktmax = K >> 5;
  const int total = 16 * ktmax * 512;
  for (int p = tid; p < total; p += nthr) {
    int tile = p >> 9;
    int r    = p & 511;
    int nt   = tile / ktmax;
    int kt   = tile - nt * ktmax;
    int lane = r >> 4;
    int jj   = r & 15;
    int hi   = lane >> 4;
    int n    = lane & 15;
    int col  = nt * 16 + n;          // column of the 256-wide gate matrix
    int k    = kt * 32 + hi * 16 + jj;
    dst[p] = (_Float16)W[col * K + k];   // B[k][col] = W[col][k]
  }
}

__global__ void prep_kernel(const float* __restrict__ Wih0, const float* __restrict__ Whh0,
                            const float* __restrict__ bih0, const float* __restrict__ bhh0,
                            const float* __restrict__ Wih1, const float* __restrict__ Whh1,
                            const float* __restrict__ bih1, const float* __restrict__ bhh1,
                            void* __restrict__ ws) {
  _Float16* wh  = (_Float16*)ws;
  float*    bs0 = (float*)((char*)ws + BSUM0_BYTE);
  float*    bs1 = (float*)((char*)ws + BSUM1_BYTE);
  int tid = threadIdx.x;
  if (tid < 256) {
    bs0[tid] = bih0[tid] + bhh0[tid];
    bs1[tid] = bih1[tid] + bhh1[tid];
  }
  pack_w(Wih0, 32, wh + PK0X_OFF, tid, blockDim.x);
  pack_w(Whh0, 64, wh + PK0H_OFF, tid, blockDim.x);
  pack_w(Wih1, 64, wh + PK1X_OFF, tid, blockDim.x);
  pack_w(Whh1, 64, wh + PK1H_OFF, tid, blockDim.x);
}

// ---------------------------------------------------------------------------
// Fast activations: sigma(y) = rcp(1 + exp2(-log2e * y)); exact saturation at
// the rails since rcp(1+inf)=0 and rcp(1+0)=1 -> no clamping required.
// tanh(x) = 2*sigma(2x) - 1.
// ---------------------------------------------------------------------------
__device__ inline float tanh_rcp(float x) {
  float e = __builtin_amdgcn_exp2f(x * (-2.0f * LOG2E));
  return fmaf(2.0f, __builtin_amdgcn_rcpf(1.0f + e), -1.0f);
}

// A operand (16xK tile, f16): lane r=l&15,hi=l>>4; halves 0..7 = K[k0+16hi ..],
// halves 8..15 = K[k0+16hi+8 ..] -> two contiguous 16B LDS loads.
__device__ inline v16h load_A16(const _Float16* __restrict__ base, int ld, int k0, int lane) {
  int r = lane & 15, hi = lane >> 4;
  v8h a0 = *(const v8h*)(base + r * ld + k0 + hi * 16);
  v8h a1 = *(const v8h*)(base + r * ld + k0 + hi * 16 + 8);
  return __builtin_shufflevector(a0, a1, 0, 1, 2, 3, 4, 5, 6, 7,
                                         8, 9, 10, 11, 12, 13, 14, 15);
}

// ---------------------------------------------------------------------------
// Fused 2-layer LSTM + head.
// Block = 16 batch rows, 256 threads = 8 waves; wave w owns 2 N-tiles
// (columns 32w .. 32w+31 of the 256-wide gate vector; gate = w>>1).
// ---------------------------------------------------------------------------
__global__ void __launch_bounds__(256)
lstm_fused_kernel(const float* __restrict__ x,
                  const float* __restrict__ W1, const float* __restrict__ b1h,
                  const float* __restrict__ W2, const float* __restrict__ b2,
                  const void* __restrict__ ws, float* __restrict__ out) {
  __shared__ _Float16 sh_x[16 * 32];     // x_t tile, A layout row-major
  __shared__ _Float16 sh_h0[16 * 64];    // layer0 hidden (f16)
  __shared__ _Float16 sh_h1[16 * 64];    // layer1 hidden (f16)
  __shared__ float    sh_g[4 * 64 * 16]; // gate exchange: [gate][col][m]
  __shared__ float    sh_red[256];

  const _Float16* wh  = (const _Float16*)ws;
  const float*    bs0 = (const float*)((const char*)ws + BSUM0_BYTE);
  const float*    bs1 = (const float*)((const char*)ws + BSUM1_BYTE);

  const int tid  = threadIdx.x;
  const int lane = tid & 31;
  const int w    = tid >> 5;        // wave id 0..7
  const int gate = w >> 1;          // i,f,g,o
  const int half = w & 1;           // which half of the gate's 64 columns
  const int n16  = lane & 15;
  const int hi   = lane >> 4;
  const int row0 = blockIdx.x * 16;

  // ---- register-resident packed B operands (14 x v16h = 112 VGPRs/wave) ----
  v16h B0x[2], B0h[4], B1x[4], B1h[4];
#pragma unroll
  for (int j = 0; j < 2; ++j) {
    const int nt = gate * 4 + half * 2 + j;
    B0x[j] = *(const v16h*)(wh + PK0X_OFF + nt * 512 + lane * 16);
#pragma unroll
    for (int kt = 0; kt < 2; ++kt) {
      B0h[j * 2 + kt] = *(const v16h*)(wh + PK0H_OFF + (nt * 2 + kt) * 512 + lane * 16);
      B1x[j * 2 + kt] = *(const v16h*)(wh + PK1X_OFF + (nt * 2 + kt) * 512 + lane * 16);
      B1h[j * 2 + kt] = *(const v16h*)(wh + PK1H_OFF + (nt * 2 + kt) * 512 + lane * 16);
    }
  }

  // branch-free activation constants (uniform per wave):
  // act(x) = sAct * sigma(sAct*(x+bias)) + oAct   (sigmoid or tanh)
  const float sAct = (gate == 2) ? 2.0f : 1.0f;
  const float oAct = (gate == 2) ? -1.0f : 0.0f;
  const float aAct = -LOG2E * sAct;      // exp2 argument scale
  float ab0[2], ab1[2];                  // aAct * bias, per owned column
#pragma unroll
  for (int j = 0; j < 2; ++j) {
    const int col = (gate * 4 + half * 2 + j) * 16 + n16;
    ab0[j] = aAct * bs0[col];
    ab1[j] = aAct * bs1[col];
  }

  // elementwise ownership: row m, 4 contiguous columns n0..n0+3
  const int m  = tid >> 4;
  const int n0 = (tid & 15) * 4;
  float c0[4], c1[4];
#pragma unroll
  for (int e = 0; e < 4; ++e) { c0[e] = 0.f; c1[e] = 0.f; }
  {
    v4h z = {};
    *(v4h*)(sh_h0 + m * 64 + n0) = z;
    *(v4h*)(sh_h1 + m * 64 + n0) = z;
  }
  __syncthreads();

  const int xr = tid >> 4;
  const int xk = (tid & 15) * 2;
  const float* xbase = x + (size_t)(row0 + xr) * SEQ * IND + xk;

  for (int t = 0; t < SEQ; ++t) {
    // ---- stage x_t (f32 -> f16 into LDS) ----
    {
      const float* xp = xbase + (size_t)t * IND;
      v2f xv = *(const v2f*)xp;
      if (t + 1 < SEQ) __builtin_prefetch(xp + IND, 0, 0);
      v2h hv;
      hv[0] = (_Float16)xv[0];
      hv[1] = (_Float16)xv[1];
      *(v2h*)(sh_x + xr * 32 + xk) = hv;
    }
    __syncthreads();

    // ================= layer 0: g = x_t@Wih0^T + h0@Whh0^T + b =================
    v8f acc[2] = {};
    {
      // hoist ALL A-tile loads so ds latency overlaps the WMMA chain
      v16h ax  = load_A16(sh_x, 32, 0, lane);
      v16h ah0 = load_A16(sh_h0, 64, 0, lane);
      v16h ah1 = load_A16(sh_h0, 64, 32, lane);
#pragma unroll
      for (int j = 0; j < 2; ++j) acc[j] = WMMA_F16(ax, B0x[j], acc[j]);
#pragma unroll
      for (int j = 0; j < 2; ++j) acc[j] = WMMA_F16(ah0, B0h[j * 2 + 0], acc[j]);
#pragma unroll
      for (int j = 0; j < 2; ++j) acc[j] = WMMA_F16(ah1, B0h[j * 2 + 1], acc[j]);
    }
    // activation (branch-free) + exchange via LDS
#pragma unroll
    for (int j = 0; j < 2; ++j) {
#pragma unroll
      for (int v = 0; v < 8; ++v) {
        float e = __builtin_amdgcn_exp2f(fmaf(aAct, acc[j][v], ab0[j]));
        acc[j][v] = fmaf(sAct, __builtin_amdgcn_rcpf(1.0f + e), oAct);
      }
      float* gp = sh_g + (gate * 64 + (half * 2 + j) * 16 + n16) * 16 + hi * 8;
      v4f lo = {acc[j][0], acc[j][1], acc[j][2], acc[j][3]};
      v4f hh = {acc[j][4], acc[j][5], acc[j][6], acc[j][7]};
      *(v4f*)gp = lo;
      *(v4f*)(gp + 4) = hh;
    }
    __syncthreads();
    {
      v4h hreg;
#pragma unroll
      for (int e = 0; e < 4; ++e) {
        int n = n0 + e;
        float iv = sh_g[0 * 1024 + n * 16 + m];
        float fv = sh_g[1 * 1024 + n * 16 + m];
        float gv = sh_g[2 * 1024 + n * 16 + m];
        float ov = sh_g[3 * 1024 + n * 16 + m];
        c0[e] = fmaf(fv, c0[e], iv * gv);
        hreg[e] = (_Float16)(ov * tanh_rcp(c0[e]));
      }
      *(v4h*)(sh_h0 + m * 64 + n0) = hreg;
    }
    __syncthreads();

    // ================= layer 1: g = h0@Wih1^T + h1@Whh1^T + b =================
#pragma unroll
    for (int j = 0; j < 2; ++j) acc[j] = (v8f){};
    {
      v16h a00 = load_A16(sh_h0, 64, 0, lane);
      v16h a01 = load_A16(sh_h0, 64, 32, lane);
      v16h a10 = load_A16(sh_h1, 64, 0, lane);
      v16h a11 = load_A16(sh_h1, 64, 32, lane);
#pragma unroll
      for (int j = 0; j < 2; ++j) acc[j] = WMMA_F16(a00, B1x[j * 2 + 0], acc[j]);
#pragma unroll
      for (int j = 0; j < 2; ++j) acc[j] = WMMA_F16(a01, B1x[j * 2 + 1], acc[j]);
#pragma unroll
      for (int j = 0; j < 2; ++j) acc[j] = WMMA_F16(a10, B1h[j * 2 + 0], acc[j]);
#pragma unroll
      for (int j = 0; j < 2; ++j) acc[j] = WMMA_F16(a11, B1h[j * 2 + 1], acc[j]);
    }
#pragma unroll
    for (int j = 0; j < 2; ++j) {
#pragma unroll
      for (int v = 0; v < 8; ++v) {
        float e = __builtin_amdgcn_exp2f(fmaf(aAct, acc[j][v], ab1[j]));
        acc[j][v] = fmaf(sAct, __builtin_amdgcn_rcpf(1.0f + e), oAct);
      }
      float* gp = sh_g + (gate * 64 + (half * 2 + j) * 16 + n16) * 16 + hi * 8;
      v4f lo = {acc[j][0], acc[j][1], acc[j][2], acc[j][3]};
      v4f hh = {acc[j][4], acc[j][5], acc[j][6], acc[j][7]};
      *(v4f*)gp = lo;
      *(v4f*)(gp + 4) = hh;
    }
    __syncthreads();
    {
      v4h hreg;
#pragma unroll
      for (int e = 0; e < 4; ++e) {
        int n = n0 + e;
        float iv = sh_g[0 * 1024 + n * 16 + m];
        float fv = sh_g[1 * 1024 + n * 16 + m];
        float gv = sh_g[2 * 1024 + n * 16 + m];
        float ov = sh_g[3 * 1024 + n * 16 + m];
        c1[e] = fmaf(fv, c1[e], iv * gv);
        hreg[e] = (_Float16)(ov * tanh_rcp(c1[e]));
      }
      *(v4h*)(sh_h1 + m * 64 + n0) = hreg;
    }
    __syncthreads();
  }

  // ================= head: out = relu(h1 @ W1^T + b1) @ W2^T + b2 =============
  {
    float partial = 0.f;
#pragma unroll
    for (int qq = 0; qq < 4; ++qq) {
      int q = n0 + qq;            // 4 neurons per thread
      float s = b1h[q];
#pragma unroll 8
      for (int n = 0; n < 64; ++n)
        s += (float)sh_h1[m * 64 + n] * W1[q * 64 + n];
      s = fmaxf(s, 0.f);
      partial += s * W2[q];
    }
    sh_red[tid] = partial;
    __syncthreads();
    if ((tid & 15) == 0) {
      float s = b2[0];
#pragma unroll
      for (int i = 0; i < 16; ++i) s += sh_red[tid + i];
      out[row0 + m] = s;
    }
  }
}

// ---------------------------------------------------------------------------
extern "C" void kernel_launch(void* const* d_in, const int* in_sizes, int n_in,
                              void* d_out, int out_size, void* d_ws, size_t ws_size,
                              hipStream_t stream) {
  const float* x    = (const float*)d_in[0];
  const float* Wih0 = (const float*)d_in[1];
  const float* Whh0 = (const float*)d_in[2];
  const float* bih0 = (const float*)d_in[3];
  const float* bhh0 = (const float*)d_in[4];
  const float* Wih1 = (const float*)d_in[5];
  const float* Whh1 = (const float*)d_in[6];
  const float* bih1 = (const float*)d_in[7];
  const float* bhh1 = (const float*)d_in[8];
  const float* W1   = (const float*)d_in[9];
  const float* b1   = (const float*)d_in[10];
  const float* W2   = (const float*)d_in[11];
  const float* b2   = (const float*)d_in[12];
  float* out = (float*)d_out;

  prep_kernel<<<1, 256, 0, stream>>>(Wih0, Whh0, bih0, bhh0,
                                     Wih1, Whh1, bih1, bhh1, d_ws);
  lstm_fused_kernel<<<BATCH / 16, 256, 0, stream>>>(x, W1, b1, W2, b2, d_ws, out);
}
